// RoPEMultiheadAttention_24206435680286
// MI455X (gfx1250) — matrix operand order repaired and checked by reading, not compiled, not run on hardware
//
#include <hip/hip_runtime.h>

// ---------------------------------------------------------------------------
// RoPE Multi-head Attention for MI455X (gfx1250), bf16 WMMA pipeline v2.1.
//   B=4, SEQ=2048, D_MODEL=1024, N_HEADS=16, HEAD_DIM=64
// All GEMM-shaped work runs on v_wmma_f32_16x16x32_bf16 (fp32 acc).
// Global->LDS staging uses gfx1250 async-to-LDS ops (ASYNCcnt) when the
// toolchain exposes them, with a synchronous fallback otherwise.
// ---------------------------------------------------------------------------

#define D_MODEL  1024
#define N_HEADS  16
#define HEAD_DIM 64
#define SEQ_LEN  2048
#define BATCH    4

typedef __bf16 bf16_t;
typedef __attribute__((ext_vector_type(16))) __bf16        v16bf;
typedef __attribute__((ext_vector_type(8)))  float         v8f;
typedef __attribute__((ext_vector_type(4)))  unsigned int  v4u;
typedef __attribute__((ext_vector_type(4)))  int           v4i;

// ---- async global->LDS (gfx1250) with portable fallback -------------------
#if __has_builtin(__builtin_amdgcn_global_load_async_to_lds_b128) && \
    __has_builtin(__builtin_amdgcn_s_wait_asynccnt)
#define HAVE_ASYNC_LDS 1
#else
#define HAVE_ASYNC_LDS 0
#endif

__device__ __forceinline__ void copy16_g2l(const bf16_t* gsrc, bf16_t* ldst) {
#if HAVE_ASYNC_LDS
    __builtin_amdgcn_global_load_async_to_lds_b128(
        (__attribute__((address_space(1))) v4i*)gsrc,
        (__attribute__((address_space(3))) v4i*)ldst,
        0, 0);
#else
    *(v4u*)ldst = *(const v4u*)gsrc;
#endif
}

template <int N>
__device__ __forceinline__ void async_waitN() {
#if HAVE_ASYNC_LDS
    __builtin_amdgcn_s_wait_asynccnt(N);
#endif
}

// ---- WMMA fragments -------------------------------------------------------
// One WMMA A/B operand: 16 bf16 per lane = 8 VGPRs.
union ABFrag { v16bf h; v4u q[2]; };

// A-matrix 16x32 (MxK), 16-bit, per CDNA5 ISA 7.12.2:
//   lanes 0-15 : row M = lane,    elems 0..7 = K 0..7,  8..15 = K 16..23
//   lanes 16-31: row M = lane-16, elems 0..7 = K 8..15, 8..15 = K 24..31
__device__ __forceinline__ ABFrag load_frag_A(const bf16_t* p, int ld, int lane) {
    ABFrag f;
    int r   = lane & 15;
    int sel = lane >> 4;
    const bf16_t* base = p + (size_t)r * ld + sel * 8;
    f.q[0] = *(const v4u*)(base);
    f.q[1] = *(const v4u*)(base + 16);
    return f;
}

// B-matrix 32x16 (KxN), 16-bit: lanes 0-15 col N, elems 0..7 = K 0..7, 8..15 = K 16..23;
// lanes 16-31 col N, elems 0..7 = K 8..15, 8..15 = K 24..31. Stored [N,K], K contiguous.
__device__ __forceinline__ ABFrag load_frag_B(const bf16_t* p, int ld, int lane) {
    ABFrag f;
    int n    = lane & 15;
    int ksel = lane >> 4;
    const bf16_t* base = p + (size_t)n * ld + ksel * 8;
    f.q[0] = *(const v4u*)(base);
    f.q[1] = *(const v4u*)(base + 16);
    return f;
}

__device__ __forceinline__ v8f wmma_bf16(v16bf a, v16bf b, v8f c) {
    return __builtin_amdgcn_wmma_f32_16x16x32_bf16(false, a, false, b,
                                                   (short)0, c, false, false);
}

// ---- fp32 -> bf16 conversion ---------------------------------------------
__global__ void cvt_f32_to_bf16(const float* __restrict__ in,
                                bf16_t* __restrict__ out, int n) {
    int i = blockIdx.x * blockDim.x + threadIdx.x;
    if (i < n) out[i] = (bf16_t)in[i];
}

// ---- GEMM: C[M,N] = A[M,K] * B[N,K]^T (+bias) -----------------------------
// 8 waves / block, block tile 128(M) x 64(N); wave tile 32x32 (2x2 WMMA).
// Double-buffered LDS staging of A (8KB) and B (4KB) tiles via async copies.
// bias_mode: 0 = none, 1 = bias[n], 2 = bias[(m / bias_div) * N + n]
__global__ void __launch_bounds__(256)
gemm_bf16_wmma(const bf16_t* __restrict__ A,
               const bf16_t* __restrict__ Bm,
               const float*  __restrict__ bias,
               int bias_mode, int bias_div,
               float*  __restrict__ Cf,
               bf16_t* __restrict__ Cb,
               int N, int K, int out_bf16) {
    __shared__ __attribute__((aligned(16))) bf16_t a_lds[2][128 * 32];
    __shared__ __attribute__((aligned(16))) bf16_t b_lds[2][64 * 32];

    int tid  = threadIdx.x;
    int lane = tid & 31;
    int w    = tid >> 5;
    int wm   = w & 3;        // 4 m-tiles of 32
    int wn   = w >> 2;       // 2 n-tiles of 32
    int m0 = blockIdx.y * 128;
    int n0 = blockIdx.x * 64;

    auto stage = [&](int k0, int buf) {
        // A tile: 128 rows x 32 k = 512 x 16B chunks, 2 per thread
#pragma unroll
        for (int j = 0; j < 2; ++j) {
            int c = tid + 256 * j;
            int row = c >> 2, ko = (c & 3) * 8;
            copy16_g2l(A + (size_t)(m0 + row) * K + k0 + ko,
                       &a_lds[buf][row * 32 + ko]);
        }
        // B tile: 64 rows x 32 k = 256 x 16B chunks, 1 per thread
        {
            int row = tid >> 2, ko = (tid & 3) * 8;
            copy16_g2l(Bm + (size_t)(n0 + row) * K + k0 + ko,
                       &b_lds[buf][row * 32 + ko]);
        }
    };

    v8f c[2][2] = {};
    const int KSTEPS = K / 32;

    stage(0, 0);
#pragma unroll 1
    for (int ks = 0; ks < KSTEPS; ++ks) {
        int cur = ks & 1;
        bool pre = (ks + 1 < KSTEPS);
        if (pre) stage((ks + 1) * 32, cur ^ 1);      // 3 async per thread
        if (pre) async_waitN<3>(); else async_waitN<0>();
        __syncthreads();

        const bf16_t* ap = &a_lds[cur][wm * 32 * 32];
        const bf16_t* bp = &b_lds[cur][wn * 32 * 32];
        ABFrag a0 = load_frag_A(ap,           32, lane);
        ABFrag a1 = load_frag_A(ap + 16 * 32, 32, lane);
        ABFrag b0 = load_frag_B(bp,           32, lane);
        ABFrag b1 = load_frag_B(bp + 16 * 32, 32, lane);
        c[0][0] = wmma_bf16(a0.h, b0.h, c[0][0]);
        c[0][1] = wmma_bf16(a0.h, b1.h, c[0][1]);
        c[1][0] = wmma_bf16(a1.h, b0.h, c[1][0]);
        c[1][1] = wmma_bf16(a1.h, b1.h, c[1][1]);
        __syncthreads();
    }

    // C/D layout: lane -> n = lane&15, VGPR r -> m = r + 8*(lane>>4)
    int nn = lane & 15, msel = lane >> 4;
#pragma unroll
    for (int mt = 0; mt < 2; ++mt)
#pragma unroll
        for (int nt = 0; nt < 2; ++nt)
#pragma unroll
            for (int r = 0; r < 8; ++r) {
                int m = m0 + wm * 32 + 16 * mt + 8 * msel + r;
                int n = n0 + wn * 32 + 16 * nt + nn;
                float v = c[mt][nt][r];
                if (bias_mode == 1)      v += bias[n];
                else if (bias_mode == 2) v += bias[(size_t)(m / bias_div) * N + n];
                if (out_bf16) Cb[(size_t)m * N + n] = (bf16_t)v;
                else          Cf[(size_t)m * N + n] = v;
            }
}

// ---- RoPE in place on bf16 [B,S,H,HEAD_DIM]; one thread per (.., d<32) ---
__global__ void rope_inplace(bf16_t* __restrict__ X, int total) {
    int i = blockIdx.x * blockDim.x + threadIdx.x;
    if (i >= total) return;
    int d = i & 31;
    int t = i >> 5;                          // (b*S + s)*H + h
    int s = (t >> 4) & (SEQ_LEN - 1);
    size_t base = (size_t)t * HEAD_DIM;
    float x1 = (float)X[base + d];
    float x2 = (float)X[base + d + 32];
    float inv = __expf(-(float)d * (6.907755278982137f / 32.0f)); // 1000^(-d/32)
    float ang = (float)s * inv;
    float sn, cs;
    __sincosf(ang, &sn, &cs);
    X[base + d]      = (bf16_t)(x1 * cs - x2 * sn);
    X[base + d + 32] = (bf16_t)(x1 * sn + x2 * cs);
}

// ---- V [B,S,H,D] -> Vt [B,H,D,S] so PV B-fragments are contiguous --------
__global__ void transpose_v(const bf16_t* __restrict__ V,
                            bf16_t* __restrict__ Vt, int total) {
    int i = blockIdx.x * blockDim.x + threadIdx.x;
    if (i >= total) return;
    int d  = i & (HEAD_DIM - 1);
    int t  = i >> 6;
    int h  = t & (N_HEADS - 1);
    int bs = t >> 4;
    int s  = bs & (SEQ_LEN - 1);
    int b  = bs >> 11;
    Vt[(((size_t)(b * N_HEADS + h) * HEAD_DIM + d) * SEQ_LEN) + s] = V[i];
}

// ---- Flash attention: 4 waves / block, 64 queries / block -----------------
// K/V tiles (32 keys) double-buffered in LDS, shared by the 4 waves.
__global__ void __launch_bounds__(128)
flash_attn_wmma(const bf16_t* __restrict__ Q,
                const bf16_t* __restrict__ Kt,
                const bf16_t* __restrict__ Vt,
                bf16_t* __restrict__ O) {
    __shared__ __attribute__((aligned(16))) bf16_t k_lds[2][32 * 64]; // [key][d]
    __shared__ __attribute__((aligned(16))) bf16_t v_lds[2][64 * 32]; // [d][key]
    __shared__ __attribute__((aligned(16))) bf16_t pshm[4][16 * 32];  // per-wave P

    int tid  = threadIdx.x;
    int lane = tid & 31;
    int w    = tid >> 5;
    int q0 = blockIdx.x * 64 + w * 16;
    int h  = blockIdx.y;
    int b  = blockIdx.z;

    const bf16_t* qp    = Q  + ((size_t)(b * SEQ_LEN + q0)) * D_MODEL + h * HEAD_DIM;
    const bf16_t* kbase = Kt + ((size_t)b * SEQ_LEN) * D_MODEL + h * HEAD_DIM;
    const bf16_t* vbase = Vt + ((size_t)(b * N_HEADS + h) * HEAD_DIM) * SEQ_LEN;

    ABFrag qa0 = load_frag_A(qp,      D_MODEL, lane);  // d 0..31
    ABFrag qa1 = load_frag_A(qp + 32, D_MODEL, lane);  // d 32..63

    auto stage = [&](int kt, int buf) {
        // K tile: 64 keys x d 0..31 = 256 x 16B chunks, 2 per thread
#pragma unroll
        for (int j = 0; j < 2; ++j) {
            int c = tid + 128 * j;
            int key = c >> 2, doff = (c & 3) * 8;
            copy16_g2l(kbase + (size_t)(kt + key) * D_MODEL + doff,
                       &k_lds[buf][key * 64 + doff]);
        }
        // V tile (transposed source): 64 d x 32 keys = 256 x 16B chunks
#pragma unroll
        for (int j = 0; j < 2; ++j) {
            int c = tid + 128 * j;
            int d = c >> 2, ko = (c & 3) * 8;
            copy16_g2l(vbase + (size_t)d * SEQ_LEN + kt + ko,
                       &v_lds[buf][d * 32 + ko]);
        }
    };

    v8f o[4] = {};
    float mrow[8], lrow[8];
#pragma unroll
    for (int r = 0; r < 8; ++r) { mrow[r] = -1e30f; lrow[r] = 0.0f; }
    const float scale = 0.125f;            // HEAD_DIM^-0.5
    int msel = lane >> 4, nn = lane & 15;

    stage(0, 0);
#pragma unroll 1
    for (int kt = 0; kt < SEQ_LEN; kt += 32) {
        int cur = (kt >> 5) & 1;
        bool pre = (kt + 32 < SEQ_LEN);
        if (pre) stage(kt + 32, cur ^ 1);            // 4 async per thread
        if (pre) async_waitN<4>(); else async_waitN<0>();
        __syncthreads();

        const bf16_t* kb = k_lds[cur];
        const bf16_t* vv = v_lds[cur];

        // ---- S = Q K^T for 32 keys (two 16x16 tiles) ----
        ABFrag k00 = load_frag_B(kb,                64, lane);
        ABFrag k01 = load_frag_B(kb + 32,           64, lane);
        ABFrag k10 = load_frag_B(kb + 16 * 64,      64, lane);
        ABFrag k11 = load_frag_B(kb + 16 * 64 + 32, 64, lane);
        v8f s0 = {}, s1 = {};
        s0 = wmma_bf16(qa0.h, k00.h, s0);
        s0 = wmma_bf16(qa1.h, k01.h, s0);
        s1 = wmma_bf16(qa0.h, k10.h, s1);
        s1 = wmma_bf16(qa1.h, k11.h, s1);

        // ---- online softmax; row stats per VGPR slot r (m = r + 8*msel) ----
        float alpha[8];
#pragma unroll
        for (int r = 0; r < 8; ++r) {
            float v0 = s0[r] * scale;
            float v1 = s1[r] * scale;
            float mx = fmaxf(v0, v1);
#pragma unroll
            for (int off = 1; off < 16; off <<= 1)
                mx = fmaxf(mx, __shfl_xor(mx, off, 32));
            float mnew = fmaxf(mrow[r], mx);
            float a    = __expf(mrow[r] - mnew);
            float p0   = __expf(v0 - mnew);
            float p1   = __expf(v1 - mnew);
            float sum  = p0 + p1;
#pragma unroll
            for (int off = 1; off < 16; off <<= 1)
                sum += __shfl_xor(sum, off, 32);
            lrow[r] = lrow[r] * a + sum;
            mrow[r] = mnew;
            alpha[r] = a;
            int m = 8 * msel + r;
            pshm[w][m * 32 + nn]      = (bf16_t)p0;
            pshm[w][m * 32 + 16 + nn] = (bf16_t)p1;
        }
#pragma unroll
        for (int t = 0; t < 4; ++t)
#pragma unroll
            for (int r = 0; r < 8; ++r) o[t][r] *= alpha[r];

        __syncthreads();   // order P ds_store -> ds_load; uniform across waves

        ABFrag pf = load_frag_A(pshm[w], 32, lane);   // P as 16x32 A-fragment
#pragma unroll
        for (int t = 0; t < 4; ++t) {
            ABFrag vf = load_frag_B(vv + (size_t)(t * 16) * 32, 32, lane);
            o[t] = wmma_bf16(pf.h, vf.h, o[t]);
        }
        __syncthreads();   // buffers free for next prefetch
    }

    // ---- normalize and store [B,S,H,D] as bf16 for the output GEMM ----
#pragma unroll
    for (int t = 0; t < 4; ++t)
#pragma unroll
        for (int r = 0; r < 8; ++r) {
            int m = q0 + 8 * msel + r;
            int d = t * 16 + nn;
            float val = o[t][r] / lrow[r];
            O[((size_t)(b * SEQ_LEN + m)) * D_MODEL + h * HEAD_DIM + d] = (bf16_t)val;
        }
}

// ---------------------------------------------------------------------------
extern "C" void kernel_launch(void* const* d_in, const int* in_sizes, int n_in,
                              void* d_out, int out_size, void* d_ws, size_t ws_size,
                              hipStream_t stream) {
    (void)in_sizes; (void)n_in; (void)out_size; (void)ws_size;

    const float* q_in  = (const float*)d_in[0];
    const float* k_in  = (const float*)d_in[1];
    const float* v_in  = (const float*)d_in[2];
    const float* cbias = (const float*)d_in[3];
    const float* W_q   = (const float*)d_in[4];
    const float* W_k   = (const float*)d_in[5];
    const float* W_v   = (const float*)d_in[6];
    const float* W_o   = (const float*)d_in[7];
    const float* b_o   = (const float*)d_in[8];
    float* out = (float*)d_out;

    const size_t MD = (size_t)BATCH * SEQ_LEN * D_MODEL;   // 8,388,608
    const size_t DD = (size_t)D_MODEL * D_MODEL;           // 1,048,576
    const int    M  = BATCH * SEQ_LEN;                     // 8192

    bf16_t* p = (bf16_t*)d_ws;
    bf16_t* qin_b = p; p += MD;
    bf16_t* kin_b = p; p += MD;
    bf16_t* vin_b = p; p += MD;
    bf16_t* Wq_b  = p; p += DD;
    bf16_t* Wk_b  = p; p += DD;
    bf16_t* Wv_b  = p; p += DD;
    bf16_t* Wo_b  = p; p += DD;
    bf16_t* Qb    = p; p += MD;   // projected Q (+bias), RoPE'd in place
    bf16_t* Kb    = p; p += MD;   // projected K, RoPE'd in place
    bf16_t* Vb    = p; p += MD;   // projected V
    bf16_t* Vt    = p; p += MD;   // V as [B,H,D,S]
    bf16_t* Ab    = p; p += MD;   // attention output [B,S,H,D]

    const int CT = 256;
    cvt_f32_to_bf16<<<dim3((MD + CT - 1) / CT), dim3(CT), 0, stream>>>(q_in, qin_b, (int)MD);
    cvt_f32_to_bf16<<<dim3((MD + CT - 1) / CT), dim3(CT), 0, stream>>>(k_in, kin_b, (int)MD);
    cvt_f32_to_bf16<<<dim3((MD + CT - 1) / CT), dim3(CT), 0, stream>>>(v_in, vin_b, (int)MD);
    cvt_f32_to_bf16<<<dim3((DD + CT - 1) / CT), dim3(CT), 0, stream>>>(W_q, Wq_b, (int)DD);
    cvt_f32_to_bf16<<<dim3((DD + CT - 1) / CT), dim3(CT), 0, stream>>>(W_k, Wk_b, (int)DD);
    cvt_f32_to_bf16<<<dim3((DD + CT - 1) / CT), dim3(CT), 0, stream>>>(W_v, Wv_b, (int)DD);
    cvt_f32_to_bf16<<<dim3((DD + CT - 1) / CT), dim3(CT), 0, stream>>>(W_o, Wo_b, (int)DD);

    // projections: X @ W^T ; block tile 128x64, 8 waves
    dim3 ggrid(D_MODEL / 64, M / 128);
    dim3 gblk(256);
    gemm_bf16_wmma<<<ggrid, gblk, 0, stream>>>(qin_b, Wq_b, cbias, 2, SEQ_LEN,
                                               nullptr, Qb, D_MODEL, D_MODEL, 1);
    gemm_bf16_wmma<<<ggrid, gblk, 0, stream>>>(kin_b, Wk_b, nullptr, 0, 1,
                                               nullptr, Kb, D_MODEL, D_MODEL, 1);
    gemm_bf16_wmma<<<ggrid, gblk, 0, stream>>>(vin_b, Wv_b, nullptr, 0, 1,
                                               nullptr, Vb, D_MODEL, D_MODEL, 1);

    const int rtot = BATCH * SEQ_LEN * N_HEADS * (HEAD_DIM / 2);
    rope_inplace<<<dim3((rtot + CT - 1) / CT), dim3(CT), 0, stream>>>(Qb, rtot);
    rope_inplace<<<dim3((rtot + CT - 1) / CT), dim3(CT), 0, stream>>>(Kb, rtot);

    transpose_v<<<dim3((MD + CT - 1) / CT), dim3(CT), 0, stream>>>(Vb, Vt, (int)MD);

    // flash attention: 64 queries/block, 4 waves
    flash_attn_wmma<<<dim3(SEQ_LEN / 64, N_HEADS, BATCH), dim3(128), 0, stream>>>(Qb, Kb, Vt, Ab);

    // output projection -> fp32 d_out
    gemm_bf16_wmma<<<ggrid, gblk, 0, stream>>>(Ab, Wo_b, b_o, 1, 1,
                                               out, nullptr, D_MODEL, D_MODEL, 0);
}
